// ResidualBlock_34600256537588
// MI455X (gfx1250) — matrix operand order repaired and checked by reading, not compiled, not run on hardware
//
#include <hip/hip_runtime.h>
#include <hip/hip_bf16.h>

// ---------------------------------------------------------------------------
// Problem constants (from reference): x [T=8, B=16, C=128, H=56, W=56] fp32
// ---------------------------------------------------------------------------
#define TT   8
#define BB   16
#define CC   128
#define HH   56
#define WW   56
#define HWSZ (HH * WW)          // 3136
#define NIMG (TT * BB)          // 128  (T folded into batch for conv)
#define NTOT (NIMG * CC * HWSZ) // 51,380,224 elements
#define NBLK 64                 // partial-reduction blocks per channel

typedef __attribute__((ext_vector_type(16))) __bf16        v16bf;
typedef __attribute__((ext_vector_type(8)))  float         v8f;
typedef __attribute__((ext_vector_type(4)))  unsigned int  v4u;

// Load a 32-byte (16 x bf16) WMMA fragment from LDS as two 16B chunks.
static __device__ inline v16bf ld_frag32(const unsigned short* p) {
    union { v4u u[2]; v16bf v; } t;
    t.u[0] = *(const v4u*)(p);
    t.u[1] = *(const v4u*)(p + 8);
    return t.v;
}

static __device__ inline unsigned short f32_to_bf16_rne(float f) {
    unsigned int u = __float_as_uint(f);
    u += 0x7fffu + ((u >> 16) & 1u);   // round to nearest even
    return (unsigned short)(u >> 16);
}

// ---------------------------------------------------------------------------
// Kernel 1: fp32 -> bf16 conversion, vectorized x8
// ---------------------------------------------------------------------------
__global__ __launch_bounds__(256)
void k_cvt_bf16(const float* __restrict__ x, unsigned short* __restrict__ o) {
    int i = (blockIdx.x * 256 + threadIdx.x) * 8;   // NTOT divisible by 2048
    float4 a = *(const float4*)(x + i);
    float4 b = *(const float4*)(x + i + 4);
    union { unsigned short s[8]; uint4 u; } t;
    t.s[0] = f32_to_bf16_rne(a.x); t.s[1] = f32_to_bf16_rne(a.y);
    t.s[2] = f32_to_bf16_rne(a.z); t.s[3] = f32_to_bf16_rne(a.w);
    t.s[4] = f32_to_bf16_rne(b.x); t.s[5] = f32_to_bf16_rne(b.y);
    t.s[6] = f32_to_bf16_rne(b.z); t.s[7] = f32_to_bf16_rne(b.w);
    *(uint4*)(o + i) = t.u;
}

// ---------------------------------------------------------------------------
// Kernel 2: weight transform  w[co][ci][kh][kw] fp32 -> wt[tap][co][ci] bf16
// (ci innermost so conv B-tiles are contiguous -> async-copyable to LDS)
// ---------------------------------------------------------------------------
__global__ __launch_bounds__(256)
void k_wtrans(const float* __restrict__ w, unsigned short* __restrict__ wt) {
    int i = blockIdx.x * 256 + threadIdx.x;     // 9*128*128 = 147456
    if (i >= 9 * CC * CC) return;
    int ci  = i & (CC - 1);
    int co  = (i >> 7) & (CC - 1);
    int tap = i >> 14;
    wt[i] = f32_to_bf16_rne(w[(co * CC + ci) * 9 + tap]);
}

// ---------------------------------------------------------------------------
// Kernel 3: per-channel partial sums (deterministic, no atomics)
// ---------------------------------------------------------------------------
__global__ __launch_bounds__(256)
void k_chan_stats(const float* __restrict__ y, float2* __restrict__ part) {
    const int c   = blockIdx.x;
    const int blk = blockIdx.y;
    const int tid = threadIdx.x;
    const int total = NIMG * HWSZ;              // 401408
    float s = 0.f, q = 0.f;
    for (int i = blk * 256 + tid; i < total; i += NBLK * 256) {
        int n = i / HWSZ;
        int p = i - n * HWSZ;
        float v = y[(n * CC + c) * HWSZ + p];
        s += v; q += v * v;
    }
    __shared__ float ls[256], lq[256];
    ls[tid] = s; lq[tid] = q;
    __syncthreads();
    for (int off = 128; off > 0; off >>= 1) {
        if (tid < off) { ls[tid] += ls[tid + off]; lq[tid] += lq[tid + off]; }
        __syncthreads();
    }
    if (tid == 0) part[c * NBLK + blk] = make_float2(ls[0], lq[0]);
}

// ---------------------------------------------------------------------------
// Kernel 4: finalize BN -> per-channel (scale, shift)
// ---------------------------------------------------------------------------
__global__ __launch_bounds__(128)
void k_bn_finalize(const float2* __restrict__ part,
                   const float* __restrict__ gamma, const float* __restrict__ beta,
                   float2* __restrict__ ss) {
    int c = threadIdx.x;                        // 128 threads
    float s = 0.f, q = 0.f;
    for (int b = 0; b < NBLK; ++b) { float2 p = part[c * NBLK + b]; s += p.x; q += p.y; }
    const float inv = 1.0f / (float)(NIMG * HWSZ);
    float mean  = s * inv;
    float var   = q * inv - mean * mean;
    float scale = gamma[c] * rsqrtf(var + 1e-5f);
    ss[c] = make_float2(scale, beta[c] - mean * scale);
}

// ---------------------------------------------------------------------------
// Kernel 5: 3x3 conv, pad 1, implicit GEMM with bf16 WMMA (f32 accumulate).
//   Block = one (image, output row): M = 64 px (W=56 padded), N = 128 ch.
//   - Wave tile 32M x 32N: 2 A frags + 2 B frags -> 4 WMMAs per 32-K chunk.
//   - Padded input row (58 px x 128 ci) staged once per kh, ci in natural
//     order (contiguous k, same as B): every fragment is one 32B LDS read.
//   - B tiles (128x128 per tap) double-buffered through the CDNA5 async
//     global->LDS engine: tap t+1 issued while tap t computes;
//     s_wait_asynccnt 0x8 retires only the older batch (in-order ASYNCcnt).
// ---------------------------------------------------------------------------
__global__ __launch_bounds__(256)
void k_conv3x3_wmma(const unsigned short* __restrict__ xb,   // [NIMG][CC][H][W] bf16
                    const unsigned short* __restrict__ wt,   // [9][CC(co)][CC(ci)] bf16
                    float* __restrict__ y)                   // [NIMG][CC][H][W] f32
{
    // rows 58..65 intentionally unstaged: only feed masked (ow>=56) outputs
    __shared__ __attribute__((aligned(16))) unsigned short a_lds[66][136];     // [px+1][ci-slot]
    __shared__ __attribute__((aligned(16))) unsigned short b_lds[2][128][136]; // double-buffered [co][ci]

    const int oh   = blockIdx.x;
    const int n    = blockIdx.y;
    const int tid  = threadIdx.x;
    const int lane = tid & 31;
    const int wid  = tid >> 5;
    const int wm2  = wid & 1;        // 32-px half of M
    const int wn4  = wid >> 1;       // 32-ch quarter of N
    const int g    = lane >> 4;      // half-wave
    const int l15  = lane & 15;

    // ---- A staging role: thread covers channel ci_s, pixel half hseg ------
    const int ci_s = tid >> 1;                    // 0..127
    const int hseg = tid & 1;                     // pixels [hseg*28, +28)
    // identity order within each 32-ci block (contiguous k, matches B)
    const int cl   = ci_s & 31;
    const int ab   = cl >> 3;
    const int abp  = ab;
    const int cpos = (ci_s & ~31) + ((abp << 3) | (cl & 7));

    // ---- B staging role: async b128 copies, [co][16B seg] -----------------
    const int seg  = tid & 15;                    // 16B segment within 256B row
    const int co0  = tid >> 4;                    // 0..15
    const unsigned b_base0 = (unsigned)(uintptr_t)&b_lds[0][0][0];
    const unsigned b_base1 = (unsigned)(uintptr_t)&b_lds[1][0][0];
    const unsigned short* wseg = wt + co0 * CC + seg * 8;

    // Issue one tap's B tile (8 async b128 copies per thread) into buf.
    auto issue_b = [&](int tap, unsigned lbase) {
        const unsigned short* wrow = wseg + tap * CC * CC;
        unsigned lds_off = lbase + (unsigned)(co0 * 136 + seg * 8) * 2;
        #pragma unroll
        for (int it = 0; it < 8; ++it) {   // co = it*16 + co0
            unsigned long long ga =
                (unsigned long long)(uintptr_t)(wrow + it * 16 * CC);
            unsigned lo = lds_off + (unsigned)(it * 16 * 136) * 2;
            asm volatile("global_load_async_to_lds_b128 %0, %1, off"
                         :: "v"(lo), "v"(ga) : "memory");
        }
    };

    v8f acc[2][2] = {};

    issue_b(0, b_base0);                          // prefetch tap 0

    for (int kh = 0; kh < 3; ++kh) {
        const int  ih     = oh + kh - 1;
        const bool row_ok = (unsigned)ih < (unsigned)HH;

        // ================= stage padded input row (once per kh) ===========
        __syncthreads();
        a_lds[hseg * 57][cpos] = 0;               // border zeros p=0 / p=57
        if (row_ok) {
            const unsigned short* xrow =
                xb + ((n * CC + ci_s) * HH + ih) * WW + hseg * 28;
            #pragma unroll
            for (int j = 0; j < 7; ++j) {
                uint2 d = *(const uint2*)(xrow + j * 4);  // 4 px
                int p = 1 + hseg * 28 + j * 4;
                a_lds[p + 0][cpos] = (unsigned short)(d.x);
                a_lds[p + 1][cpos] = (unsigned short)(d.x >> 16);
                a_lds[p + 2][cpos] = (unsigned short)(d.y);
                a_lds[p + 3][cpos] = (unsigned short)(d.y >> 16);
            }
        } else {
            #pragma unroll
            for (int j = 0; j < 28; ++j)
                a_lds[1 + hseg * 28 + j][cpos] = 0;
        }
        __syncthreads();

        for (int kw = 0; kw < 3; ++kw) {
            const int tap = kh * 3 + kw;
            const int cur = tap & 1;

            // All waves finished reading buf[1-cur] (previous tap): safe to
            // overwrite it with tap+1 while computing tap from buf[cur].
            __syncthreads();
            if (tap < 8) {
                issue_b(tap + 1, cur ? b_base0 : b_base1);
                asm volatile("s_wait_asynccnt 0x8" ::: "memory");  // retire tap's batch
            } else {
                asm volatile("s_wait_asynccnt 0x0" ::: "memory");
            }
            __syncthreads();

            // ============ 16 WMMAs per wave for this tap ===================
            const unsigned short* arow0 = &a_lds[wm2 * 32 + l15 + kw][g * 16];
            const unsigned short* arow1 = arow0 + 16 * 136;
            const unsigned short* brow0 = &b_lds[cur][wn4 * 32 + l15][g * 16];
            const unsigned short* brow1 = brow0 + 16 * 136;
            #pragma unroll
            for (int kc = 0; kc < CC; kc += 32) {
                v16bf a0 = ld_frag32(arow0 + kc);
                v16bf a1 = ld_frag32(arow1 + kc);
                v16bf b0 = ld_frag32(brow0 + kc);
                v16bf b1 = ld_frag32(brow1 + kc);
                acc[0][0] = __builtin_amdgcn_wmma_f32_16x16x32_bf16(
                    false, a0, false, b0, (short)0, acc[0][0], false, false);
                acc[0][1] = __builtin_amdgcn_wmma_f32_16x16x32_bf16(
                    false, a0, false, b1, (short)0, acc[0][1], false, false);
                acc[1][0] = __builtin_amdgcn_wmma_f32_16x16x32_bf16(
                    false, a1, false, b0, (short)0, acc[1][0], false, false);
                acc[1][1] = __builtin_amdgcn_wmma_f32_16x16x32_bf16(
                    false, a1, false, b1, (short)0, acc[1][1], false, false);
            }
        }
    }

    // ---- epilogue: D layout is N=lane%16, M = r + 8*(lane/16) -------------
    float* yout = y + (n * CC) * HWSZ + oh * WW;
    #pragma unroll
    for (int ma = 0; ma < 2; ++ma) {
        #pragma unroll
        for (int nb = 0; nb < 2; ++nb) {
            int co = wn4 * 32 + nb * 16 + l15;
            #pragma unroll
            for (int r = 0; r < 8; ++r) {
                int ow = wm2 * 32 + ma * 16 + g * 8 + r;
                if (ow < WW) yout[co * HWSZ + ow] = acc[ma][nb][r];
            }
        }
    }
}

// ---------------------------------------------------------------------------
// Kernel 6: fused bn1 + LIF scan over T -> spikes as bf16 (conv2 input)
// ---------------------------------------------------------------------------
__global__ __launch_bounds__(256)
void k_bn_lif_bf16(const float* __restrict__ y, const float2* __restrict__ ss,
                   unsigned short* __restrict__ sp) {
    int idx = blockIdx.x * 256 + threadIdx.x;          // over B*C*HW
    if (idx >= BB * CC * HWSZ) return;
    int c = (idx / HWSZ) & (CC - 1);
    float2 sc = ss[c];
    const int stride = BB * CC * HWSZ;
    float mem = 0.f, spike = 0.f;
    #pragma unroll
    for (int t = 0; t < TT; ++t) {
        float v = y[idx + t * stride] * sc.x + sc.y;
        mem   = mem * 0.5f * (1.f - spike) + v;
        spike = (mem > 0.5f) ? 1.f : 0.f;
        sp[idx + t * stride] = (mem > 0.5f) ? (unsigned short)0x3F80 : (unsigned short)0;
    }
}

// ---------------------------------------------------------------------------
// Kernel 7: fused bn2 + bn3(x) residual + LIF scan -> final spikes (fp32)
// ---------------------------------------------------------------------------
__global__ __launch_bounds__(256)
void k_bn_res_lif(const float* __restrict__ y2, const float* __restrict__ x,
                  const float2* __restrict__ ss2, const float2* __restrict__ ss3,
                  float* __restrict__ out) {
    int idx = blockIdx.x * 256 + threadIdx.x;
    if (idx >= BB * CC * HWSZ) return;
    int c = (idx / HWSZ) & (CC - 1);
    float2 s2 = ss2[c];
    float2 s3 = ss3[c];
    const int stride = BB * CC * HWSZ;
    float mem = 0.f, spike = 0.f;
    #pragma unroll
    for (int t = 0; t < TT; ++t) {
        float v = y2[idx + t * stride] * s2.x + s2.y
                + x [idx + t * stride] * s3.x + s3.y;
        mem   = mem * 0.5f * (1.f - spike) + v;
        spike = (mem > 0.5f) ? 1.f : 0.f;
        out[idx + t * stride] = spike;
    }
}

// ---------------------------------------------------------------------------
// Host launcher
// ---------------------------------------------------------------------------
extern "C" void kernel_launch(void* const* d_in, const int* in_sizes, int n_in,
                              void* d_out, int out_size, void* d_ws, size_t ws_size,
                              hipStream_t stream) {
    const float* x  = (const float*)d_in[0];
    const float* w1 = (const float*)d_in[1];
    const float* w2 = (const float*)d_in[2];
    const float* g1 = (const float*)d_in[3];
    const float* b1 = (const float*)d_in[4];
    const float* g2 = (const float*)d_in[5];
    const float* b2 = (const float*)d_in[6];
    const float* g3 = (const float*)d_in[7];
    const float* b3 = (const float*)d_in[8];
    float* out = (float*)d_out;

    // Workspace layout (bytes)
    char* ws = (char*)d_ws;
    const size_t SZ_BF16 = (size_t)NTOT * 2;        // 102,760,448
    const size_t SZ_F32  = (size_t)NTOT * 4;        // 205,520,896
    const size_t SZ_WT   = (size_t)9 * CC * CC * 2; // 294,912
    unsigned short* xb   = (unsigned short*)(ws);
    unsigned short* s1   = (unsigned short*)(ws + SZ_BF16);
    float*          ybuf = (float*)         (ws + 2 * SZ_BF16);
    unsigned short* w1t  = (unsigned short*)(ws + 2 * SZ_BF16 + SZ_F32);
    unsigned short* w2t  = (unsigned short*)(ws + 2 * SZ_BF16 + SZ_F32 + SZ_WT);
    float2*         part = (float2*)        (ws + 2 * SZ_BF16 + SZ_F32 + 2 * SZ_WT);
    float2*         ss1  = part + CC * NBLK;
    float2*         ss2  = ss1 + CC;
    float2*         ss3  = ss2 + CC;

    const dim3 cgrid(HH, NIMG);        // one block per (output row, image)
    const dim3 sgrid(CC, NBLK);
    const int  site_blocks = (BB * CC * HWSZ) / 256;   // 25,088

    // 1) convert x to bf16; transform weights
    k_cvt_bf16<<<NTOT / 2048, 256, 0, stream>>>(x, xb);
    k_wtrans<<<(9 * CC * CC + 255) / 256, 256, 0, stream>>>(w1, w1t);
    k_wtrans<<<(9 * CC * CC + 255) / 256, 256, 0, stream>>>(w2, w2t);

    // 2) bn3 stats on raw x (residual path)
    k_chan_stats<<<sgrid, 256, 0, stream>>>(x, part);
    k_bn_finalize<<<1, 128, 0, stream>>>(part, g3, b3, ss3);

    // 3) conv1 -> ybuf
    k_conv3x3_wmma<<<cgrid, 256, 0, stream>>>(xb, w1t, ybuf);

    // 4) bn1 stats + fused bn1+LIF -> spikes s1 (bf16)
    k_chan_stats<<<sgrid, 256, 0, stream>>>(ybuf, part);
    k_bn_finalize<<<1, 128, 0, stream>>>(part, g1, b1, ss1);
    k_bn_lif_bf16<<<site_blocks, 256, 0, stream>>>(ybuf, ss1, s1);

    // 5) conv2 -> ybuf (reuse)
    k_conv3x3_wmma<<<cgrid, 256, 0, stream>>>(s1, w2t, ybuf);

    // 6) bn2 stats + fused bn2 + bn3(x) residual + LIF -> out
    k_chan_stats<<<sgrid, 256, 0, stream>>>(ybuf, part);
    k_bn_finalize<<<1, 128, 0, stream>>>(part, g2, b2, ss2);
    k_bn_res_lif<<<site_blocks, 256, 0, stream>>>(ybuf, x, ss2, ss3, out);
}
